// DiTeBlock_4715874091337
// MI455X (gfx1250) — matrix operand order repaired and checked by reading, not compiled, not run on hardware
//
#include <hip/hip_runtime.h>
#include <math.h>

typedef __attribute__((ext_vector_type(16))) _Float16 v16h;
typedef __attribute__((ext_vector_type(8)))  float    v8f;

constexpr int B_    = 2;
constexpr int N_    = 384;
constexpr int H_    = 256;
constexpr int D_    = 128;
constexpr int MIN_  = 2 * H_ + D_;   // 640
constexpr int FH_   = 1024;
constexpr int SIXH  = 6 * H_;        // 1536
constexpr int NHEAD = 8;
constexpr int DHEAD = 32;
constexpr int KT1_  = MIN_ / 32;     // 20 k-tiles for edge GEMM
constexpr int NT1_  = H_ / 16;       // 16 n-tiles
constexpr int LDA_  = MIN_ + 8;      // padded LDS row stride (f16 units)
constexpr int ROWS_ = B_ * N_;       // 768 activation rows

__device__ __forceinline__ float silu_f(float x) { return x / (1.f + __expf(-x)); }

// ---------------------------------------------------------------------------
// 1) adaLN modulation: ada[b, 0:6H] = silu(t_emb[b]) @ W_ada.T + b_ada
// ---------------------------------------------------------------------------
__global__ void __launch_bounds__(256)
k_ada(const float* __restrict__ t_emb, const float* __restrict__ W_ada,
      const float* __restrict__ b_ada, float* __restrict__ ada)
{
    int idx = blockIdx.x * 256 + threadIdx.x;
    if (idx >= B_ * SIXH) return;
    int b = idx / SIXH, r = idx - b * SIXH;
    const float* t = t_emb + b * H_;
    const float* w = W_ada + (size_t)r * H_;
    float s = b_ada[r];
    for (int k = 0; k < H_; ++k) s += silu_f(t[k]) * w[k];
    ada[idx] = s;
}

// ---------------------------------------------------------------------------
// 2) x_norm = modulate(ln(x*mask), shift_msa, scale_msa)  -> f16
// ---------------------------------------------------------------------------
__global__ void __launch_bounds__(256)
k_xnorm16(const float* __restrict__ x, const float* __restrict__ mask,
          const float* __restrict__ ada, _Float16* __restrict__ xn16)
{
    __shared__ float r1[256], r2[256];
    int row = blockIdx.x;
    int b = row / N_;
    int tid = threadIdx.x;
    float mv = mask[row];
    float v = x[(size_t)row * H_ + tid] * mv;
    r1[tid] = v; r2[tid] = v * v;
    __syncthreads();
    for (int off = 128; off > 0; off >>= 1) {
        if (tid < off) { r1[tid] += r1[tid + off]; r2[tid] += r2[tid + off]; }
        __syncthreads();
    }
    float mean = r1[0] * (1.f / H_);
    float var  = r2[0] * (1.f / H_) - mean * mean;
    float y = (v - mean) * rsqrtf(var + 1e-6f);
    float sh = ada[b * SIXH + tid];
    float sc = ada[b * SIXH + H_ + tid];
    xn16[(size_t)row * H_ + tid] = (_Float16)(y * (1.f + sc) + sh);
}

// ---------------------------------------------------------------------------
// 3) generic weight pack: W (Ndim x Kdim row-major f32) -> WMMA-B fragment
//    order f16: Bf[kt][nt][lane][e], k = kt*32+8*(lane/16)+(e%8)+16*(e/8), n = nt*16+lane%16
// ---------------------------------------------------------------------------
__global__ void __launch_bounds__(256)
k_packB(const float* __restrict__ W, _Float16* __restrict__ Bf,
        int Kdim, int NTdim, int total)
{
    int idx = blockIdx.x * 256 + threadIdx.x;
    if (idx >= total) return;
    int e    = idx & 15;
    int lane = (idx >> 4) & 31;
    int rest = idx >> 9;
    int nt   = rest % NTdim;
    int kt   = rest / NTdim;
    int k = kt * 32 + 8 * (lane >> 4) + (e & 7) + 16 * (e >> 3);
    int n = nt * 16 + (lane & 15);
    Bf[idx] = (_Float16)W[(size_t)n * Kdim + k];
}

// ---------------------------------------------------------------------------
// A-fragment gather from an LDS tile row (ISA 16-bit A layout)
// ---------------------------------------------------------------------------
__device__ __forceinline__ v16h load_afrag(const _Float16* rowBase, int kbase, int hh)
{
    const unsigned int* rowp = (const unsigned int*)rowBase;
    alignas(32) unsigned int au[8];
    int u0 = (kbase + 8 * hh) >> 1;
    int u1 = (kbase + 16 + 8 * hh) >> 1;
    au[0] = rowp[u0];     au[1] = rowp[u0 + 1];
    au[2] = rowp[u0 + 2]; au[3] = rowp[u0 + 3];
    au[4] = rowp[u1];     au[5] = rowp[u1 + 1];
    au[6] = rowp[u1 + 2]; au[7] = rowp[u1 + 3];
    return *(const v16h*)au;
}

// ---------------------------------------------------------------------------
// 4) fused edge MLP: first GEMM + SiLU + column-sum over j.
//    grid = B*N blocks (one per (b,i)); 256 threads = 8 waves.
//    M-chunk = 32 j-rows so each B fragment pair feeds 4 WMMAs.
// ---------------------------------------------------------------------------
__global__ void __launch_bounds__(256)
k_edge(const _Float16* __restrict__ xn16, const float* __restrict__ dist,
       const float* __restrict__ emask, const _Float16* __restrict__ W1f,
       const float* __restrict__ b_fe1, float* __restrict__ Sbuf)
{
    __shared__ _Float16 tileA[32 * LDA_];   // 41.5 KB

    const int row  = blockIdx.x;      // b*N + i
    const int b    = row / N_;
    const int tid  = threadIdx.x;
    const int lane = tid & 31;
    const int wid  = tid >> 5;
    const int hh   = lane >> 4;
    const int ln16 = lane & 15;

    const int nt0 = wid * 2, nt1 = wid * 2 + 1;
    const float bias0 = b_fe1[nt0 * 16 + ln16];
    const float bias1 = b_fe1[nt1 * 16 + ln16];

    const _Float16* xi      = xn16 + (size_t)row * H_;
    const float*    distrow = dist  + (size_t)row * N_ * D_;
    const float*    emrow   = emask + (size_t)row * N_;

    float srun0 = 0.f, srun1 = 0.f;

    for (int jc = 0; jc < N_ / 32; ++jc) {   // 12 chunks of 32 j-rows
        __syncthreads();
        for (int idx = tid; idx < 32 * MIN_; idx += 256) {
            int r = idx / MIN_;
            int k = idx - r * MIN_;
            int j = jc * 32 + r;
            float em = emrow[j];
            float val;
            if (k < H_)            val = (float)xi[k];
            else if (k < 2 * H_)   val = (float)xn16[((size_t)b * N_ + j) * H_ + (k - H_)];
            else                   val = distrow[(size_t)j * D_ + (k - 2 * H_)];
            tileA[r * LDA_ + k] = (_Float16)(val * em);
        }
        __syncthreads();

        v8f a00 = {bias0, bias0, bias0, bias0, bias0, bias0, bias0, bias0};
        v8f a01 = {bias1, bias1, bias1, bias1, bias1, bias1, bias1, bias1};
        v8f a10 = a00, a11 = a01;

        const _Float16* r0 = tileA + ln16 * LDA_;          // m-tile 0 rows
        const _Float16* r1 = tileA + (16 + ln16) * LDA_;   // m-tile 1 rows
        for (int kt = 0; kt < KT1_; ++kt) {
            v16h bf0 = *(const v16h*)(W1f + (((size_t)(kt * NT1_ + nt0) * 32) + lane) * 16);
            v16h bf1 = *(const v16h*)(W1f + (((size_t)(kt * NT1_ + nt1) * 32) + lane) * 16);
            v16h af0 = load_afrag(r0, kt * 32, hh);
            v16h af1 = load_afrag(r1, kt * 32, hh);
            a00 = __builtin_amdgcn_wmma_f32_16x16x32_f16(false, af0, false, bf0, (short)0, a00, false, false);
            a01 = __builtin_amdgcn_wmma_f32_16x16x32_f16(false, af0, false, bf1, (short)0, a01, false, false);
            a10 = __builtin_amdgcn_wmma_f32_16x16x32_f16(false, af1, false, bf0, (short)0, a10, false, false);
            a11 = __builtin_amdgcn_wmma_f32_16x16x32_f16(false, af1, false, bf1, (short)0, a11, false, false);
        }

        float p0 = 0.f, p1 = 0.f;
#pragma unroll
        for (int e = 0; e < 8; ++e) {
            p0 += silu_f(a00[e]) + silu_f(a10[e]);
            p1 += silu_f(a01[e]) + silu_f(a11[e]);
        }
        p0 += __shfl_xor(p0, 16, 32);
        p1 += __shfl_xor(p1, 16, 32);
        srun0 += p0; srun1 += p1;
    }

    if (lane < 16) {
        Sbuf[(size_t)row * H_ + nt0 * 16 + ln16] = srun0;
        Sbuf[(size_t)row * H_ + nt1 * 16 + ln16] = srun1;
    }
}

// ---------------------------------------------------------------------------
// 5) generic WMMA GEMM: C(768 x Ndim) = A(768 x Kdim, f32) @ Bf(packed)^T
//    block = 32 rows x 256 cols (8 waves, each 2x2 WMMA accumulators).
//    EPI: 0 plain | 1 +biasScale*bias[n] | 2 res+gate*acc | 3 (res+gate*acc)*mask
// ---------------------------------------------------------------------------
template<int EPI>
__global__ void __launch_bounds__(256)
k_gemm(const float* __restrict__ A, const _Float16* __restrict__ Bf,
       float* __restrict__ Cout, int Kdim, int Ndim,
       const float* __restrict__ bias, float biasScale,
       const float* __restrict__ res, const float* __restrict__ ada,
       const float* __restrict__ mask, int gateOff)
{
    __shared__ _Float16 sA[32 * 40];   // 32 rows x 32 k, stride 40

    const int tid  = threadIdx.x;
    const int lane = tid & 31;
    const int wid  = tid >> 5;
    const int hh   = lane >> 4;
    const int ln16 = lane & 15;

    const int mbase = blockIdx.x * 32;
    const int NTdim = Ndim >> 4;
    const int ntg0  = blockIdx.y * 16 + wid * 2;
    const int ntg1  = ntg0 + 1;

    v8f a00 = {0,0,0,0,0,0,0,0};
    v8f a01 = a00, a10 = a00, a11 = a00;

    const int KT = Kdim >> 5;
    for (int kt = 0; kt < KT; ++kt) {
        __syncthreads();
        for (int s = tid; s < 32 * 32; s += 256) {
            int r = s >> 5, kl = s & 31;
            sA[r * 40 + kl] = (_Float16)A[(size_t)(mbase + r) * Kdim + kt * 32 + kl];
        }
        __syncthreads();

        v16h bf0 = *(const v16h*)(Bf + (((size_t)(kt * NTdim + ntg0) * 32) + lane) * 16);
        v16h bf1 = *(const v16h*)(Bf + (((size_t)(kt * NTdim + ntg1) * 32) + lane) * 16);
        v16h af0 = load_afrag(sA + ln16 * 40, 0, hh);
        v16h af1 = load_afrag(sA + (16 + ln16) * 40, 0, hh);
        a00 = __builtin_amdgcn_wmma_f32_16x16x32_f16(false, af0, false, bf0, (short)0, a00, false, false);
        a01 = __builtin_amdgcn_wmma_f32_16x16x32_f16(false, af0, false, bf1, (short)0, a01, false, false);
        a10 = __builtin_amdgcn_wmma_f32_16x16x32_f16(false, af1, false, bf0, (short)0, a10, false, false);
        a11 = __builtin_amdgcn_wmma_f32_16x16x32_f16(false, af1, false, bf1, (short)0, a11, false, false);
    }

    // epilogue: scatter 8 elements per accumulator
#pragma unroll
    for (int mt = 0; mt < 2; ++mt) {
#pragma unroll
        for (int ntl = 0; ntl < 2; ++ntl) {
            v8f acc = (mt == 0) ? (ntl == 0 ? a00 : a01) : (ntl == 0 ? a10 : a11);
            int col = (ntl == 0 ? ntg0 : ntg1) * 16 + ln16;
#pragma unroll
            for (int e = 0; e < 8; ++e) {
                int rowg = mbase + mt * 16 + 8 * hh + e;
                float v = acc[e];
                if constexpr (EPI == 1) {
                    v += biasScale * bias[col];
                } else if constexpr (EPI == 2) {
                    float gate = ada[(rowg / N_) * SIXH + gateOff + col];
                    v = res[(size_t)rowg * Ndim + col] + gate * v;
                } else if constexpr (EPI == 3) {
                    float gate = ada[(rowg / N_) * SIXH + gateOff + col];
                    v = (res[(size_t)rowg * Ndim + col] + gate * v) * mask[rowg];
                }
                Cout[(size_t)rowg * Ndim + col] = v;
            }
        }
    }
}

// ---------------------------------------------------------------------------
// 6) Q/K layernorm over full H (in-place on qkv)
// ---------------------------------------------------------------------------
__global__ void __launch_bounds__(256)
k_qkln(float* __restrict__ qkv, const float* __restrict__ mask)
{
    __shared__ float r1[256], r2[256];
    int bx = blockIdx.x;
    int row = bx >> 1, part = bx & 1;
    int tid = threadIdx.x;
    float* base = qkv + (size_t)row * 3 * H_ + part * H_;
    float mv = mask[row];
    float v = base[tid] * mv;
    r1[tid] = v; r2[tid] = v * v;
    __syncthreads();
    for (int off = 128; off > 0; off >>= 1) {
        if (tid < off) { r1[tid] += r1[tid + off]; r2[tid] += r2[tid + off]; }
        __syncthreads();
    }
    float mean = r1[0] * (1.f / H_);
    float var  = r2[0] * (1.f / H_) - mean * mean;
    base[tid] = (v - mean) * rsqrtf(var + 1e-6f);
}

// ---------------------------------------------------------------------------
// 7) attention: one thread per (b,h,q); online softmax over 384 keys
// ---------------------------------------------------------------------------
__global__ void __launch_bounds__(128)
k_attn(const float* __restrict__ qkv, const float* __restrict__ mask,
       float* __restrict__ ao)
{
    int bh = blockIdx.x;
    int b = bh / NHEAD, h = bh - b * NHEAD;
    int qi = blockIdx.y * 128 + threadIdx.x;
    if (qi >= N_) return;

    const float scale = 0.17677669529663689f;
    const float* Qp = qkv + ((size_t)(b * N_ + qi) * 3 * H_) + h * DHEAD;
    float qv[DHEAD];
#pragma unroll
    for (int d = 0; d < DHEAD; ++d) qv[d] = Qp[d];

    float mx = -1e30f, den = 0.f;
    float accv[DHEAD];
#pragma unroll
    for (int d = 0; d < DHEAD; ++d) accv[d] = 0.f;

    for (int j = 0; j < N_; ++j) {
        const float* Kp = qkv + ((size_t)(b * N_ + j) * 3 * H_) + H_ + h * DHEAD;
        const float* Vp = qkv + ((size_t)(b * N_ + j) * 3 * H_) + 2 * H_ + h * DHEAD;
        float dot = 0.f;
#pragma unroll
        for (int d = 0; d < DHEAD; ++d) dot += qv[d] * Kp[d];
        float sc = dot * scale + (mask[b * N_ + j] > 0.f ? 0.f : -1e9f);
        float nm = fmaxf(mx, sc);
        float c  = __expf(mx - nm);
        float p  = __expf(sc - nm);
        den = den * c + p;
#pragma unroll
        for (int d = 0; d < DHEAD; ++d) accv[d] = accv[d] * c + p * Vp[d];
        mx = nm;
    }
    float inv = 1.f / den;
    float* aop = ao + (size_t)(b * N_ + qi) * H_ + h * DHEAD;
#pragma unroll
    for (int d = 0; d < DHEAD; ++d) aop[d] = accv[d] * inv;
}

// ---------------------------------------------------------------------------
// 8) xm = modulate(ln(xr*mask), shift_mlp, scale_mlp)
// ---------------------------------------------------------------------------
__global__ void __launch_bounds__(256)
k_xm(const float* __restrict__ xr, const float* __restrict__ mask,
     const float* __restrict__ ada, float* __restrict__ xm)
{
    __shared__ float r1[256], r2[256];
    int row = blockIdx.x;
    int b = row / N_;
    int tid = threadIdx.x;
    float mv = mask[row];
    float v = xr[(size_t)row * H_ + tid] * mv;
    r1[tid] = v; r2[tid] = v * v;
    __syncthreads();
    for (int off = 128; off > 0; off >>= 1) {
        if (tid < off) { r1[tid] += r1[tid + off]; r2[tid] += r2[tid + off]; }
        __syncthreads();
    }
    float mean = r1[0] * (1.f / H_);
    float var  = r2[0] * (1.f / H_) - mean * mean;
    float y = (v - mean) * rsqrtf(var + 1e-6f);
    float sh = ada[b * SIXH + 3 * H_ + tid];
    float sc = ada[b * SIXH + 4 * H_ + tid];
    xm[(size_t)row * H_ + tid] = y * (1.f + sc) + sh;
}

// ---------------------------------------------------------------------------
// 9) g = silu(h_a) * h_b
// ---------------------------------------------------------------------------
__global__ void __launch_bounds__(256)
k_swiglu(const float* __restrict__ h12, float* __restrict__ g)
{
    int idx = blockIdx.x * 256 + threadIdx.x;
    if (idx >= ROWS_ * FH_) return;
    int row = idx / FH_, f = idx - row * FH_;
    float ha = h12[(size_t)row * 2 * FH_ + f];
    float hb = h12[(size_t)row * 2 * FH_ + FH_ + f];
    g[idx] = silu_f(ha) * hb;
}

// ---------------------------------------------------------------------------
extern "C" void kernel_launch(void* const* d_in, const int* in_sizes, int n_in,
                              void* d_out, int out_size, void* d_ws, size_t ws_size,
                              hipStream_t stream)
{
    const float* mask   = (const float*)d_in[0];
    const float* x      = (const float*)d_in[1];
    const float* t_emb  = (const float*)d_in[2];
    const float* dist   = (const float*)d_in[3];
    const float* emask  = (const float*)d_in[4];
    const float* W_ada  = (const float*)d_in[5];
    const float* b_ada  = (const float*)d_in[6];
    const float* W_fe1  = (const float*)d_in[7];
    const float* b_fe1  = (const float*)d_in[8];
    const float* W_fe2  = (const float*)d_in[9];
    const float* b_fe2  = (const float*)d_in[10];
    const float* W_qkv  = (const float*)d_in[11];
    const float* W_out  = (const float*)d_in[12];
    const float* W_ffn1 = (const float*)d_in[13];
    const float* W_ffn2 = (const float*)d_in[14];
    float* out = (float*)d_out;

    char* p = (char*)d_ws;
    auto alloc = [&](size_t bytes) -> char* {
        char* r = p;
        p += (bytes + 255) & ~(size_t)255;
        return r;
    };
    float*    ada    = (float*)   alloc((size_t)B_ * SIXH * 4);
    _Float16* xn16   = (_Float16*)alloc((size_t)ROWS_ * H_ * 2);
    _Float16* W1f    = (_Float16*)alloc((size_t)MIN_ * H_ * 2);
    _Float16* W2f    = (_Float16*)alloc((size_t)H_ * H_ * 2);
    _Float16* Wqkvf  = (_Float16*)alloc((size_t)3 * H_ * H_ * 2);
    _Float16* Woutf  = (_Float16*)alloc((size_t)H_ * H_ * 2);
    _Float16* Wffn1f = (_Float16*)alloc((size_t)2 * FH_ * H_ * 2);
    _Float16* Wffn2f = (_Float16*)alloc((size_t)H_ * FH_ * 2);
    float* Sbuf = (float*)alloc((size_t)ROWS_ * H_ * 4);
    float* xmsg = (float*)alloc((size_t)ROWS_ * H_ * 4);
    float* qkv  = (float*)alloc((size_t)ROWS_ * 3 * H_ * 4);
    float* ao   = (float*)alloc((size_t)ROWS_ * H_ * 4);
    float* xr   = (float*)alloc((size_t)ROWS_ * H_ * 4);
    float* xm   = (float*)alloc((size_t)ROWS_ * H_ * 4);
    float* h12  = (float*)alloc((size_t)ROWS_ * 2 * FH_ * 4);
    float* g    = (float*)alloc((size_t)ROWS_ * FH_ * 4);
    (void)in_sizes; (void)n_in; (void)out_size; (void)ws_size;

    auto cdiv = [](int a, int b) { return (a + b - 1) / b; };

    k_ada    <<<cdiv(B_ * SIXH, 256), 256, 0, stream>>>(t_emb, W_ada, b_ada, ada);
    k_xnorm16<<<ROWS_, 256, 0, stream>>>(x, mask, ada, xn16);

    // weight packs (fragment order)
    k_packB<<<cdiv(MIN_ * H_, 256), 256, 0, stream>>>(W_fe1, W1f, MIN_, NT1_, MIN_ * H_);
    k_packB<<<cdiv(H_ * H_, 256), 256, 0, stream>>>(W_fe2, W2f, H_, H_ / 16, H_ * H_);
    k_packB<<<cdiv(3 * H_ * H_, 256), 256, 0, stream>>>(W_qkv, Wqkvf, H_, 3 * H_ / 16, 3 * H_ * H_);
    k_packB<<<cdiv(H_ * H_, 256), 256, 0, stream>>>(W_out, Woutf, H_, H_ / 16, H_ * H_);
    k_packB<<<cdiv(2 * FH_ * H_, 256), 256, 0, stream>>>(W_ffn1, Wffn1f, H_, 2 * FH_ / 16, 2 * FH_ * H_);
    k_packB<<<cdiv(H_ * FH_, 256), 256, 0, stream>>>(W_ffn2, Wffn2f, FH_, H_ / 16, H_ * FH_);

    // fused edge MLP (dominant): Sbuf[b,i,:] = sum_j silu(feats @ W_fe1.T + b_fe1)
    k_edge<<<ROWS_, 256, 0, stream>>>(xn16, dist, emask, W1f, b_fe1, Sbuf);

    // x_msg = Sbuf @ W_fe2.T + N*b_fe2
    k_gemm<1><<<dim3(ROWS_ / 32, 1), 256, 0, stream>>>(Sbuf, W2f, xmsg, H_, H_,
            b_fe2, (float)N_, nullptr, nullptr, nullptr, 0);
    // qkv = x_msg @ W_qkv.T
    k_gemm<0><<<dim3(ROWS_ / 32, 3), 256, 0, stream>>>(xmsg, Wqkvf, qkv, H_, 3 * H_,
            nullptr, 0.f, nullptr, nullptr, nullptr, 0);
    k_qkln<<<2 * ROWS_, 256, 0, stream>>>(qkv, mask);
    k_attn<<<dim3(B_ * NHEAD, cdiv(N_, 128)), 128, 0, stream>>>(qkv, mask, ao);
    // xr = x + gate_msa * (ao @ W_out.T)
    k_gemm<2><<<dim3(ROWS_ / 32, 1), 256, 0, stream>>>(ao, Woutf, xr, H_, H_,
            nullptr, 0.f, x, ada, nullptr, 2 * H_);
    k_xm<<<ROWS_, 256, 0, stream>>>(xr, mask, ada, xm);
    // h12 = xm @ W_ffn1.T
    k_gemm<0><<<dim3(ROWS_ / 32, 8), 256, 0, stream>>>(xm, Wffn1f, h12, H_, 2 * FH_,
            nullptr, 0.f, nullptr, nullptr, nullptr, 0);
    k_swiglu<<<cdiv(ROWS_ * FH_, 256), 256, 0, stream>>>(h12, g);
    // out = (xr + gate_mlp * (g @ W_ffn2.T)) * mask
    k_gemm<3><<<dim3(ROWS_ / 32, 1), 256, 0, stream>>>(g, Wffn2f, out, FH_, H_,
            nullptr, 0.f, xr, ada, mask, 5 * H_);
}
